// NF4_56581899157525
// MI455X (gfx1250) — matrix operand order repaired and checked by reading, not compiled, not run on hardware
//
#include <hip/hip_runtime.h>

// ---------------------------------------------------------------------------
// NF4-dequant + GEMM for MI455X (gfx1250, wave32, WMMA).
//   out[8192,4096] = x[8192,4096] @ dequant(Wq)[4096,4096] + bias
// Strategy: dequant NF4 -> bf16 W_T (N-major) once in ws, convert x -> bf16,
// then a 128x128x32-tiled bf16 WMMA GEMM with f32 accumulation, using
// double-buffered LDS fed by GLOBAL_LOAD_ASYNC_TO_LDS_B128 (ASYNCcnt).
// ---------------------------------------------------------------------------

typedef __bf16 bf16;
typedef __attribute__((ext_vector_type(16))) __bf16 v16bf;
typedef __attribute__((ext_vector_type(8)))  __bf16 v8bf;
typedef __attribute__((ext_vector_type(8)))  float  v8f;
typedef __attribute__((ext_vector_type(4)))  float  v4f;
typedef __attribute__((ext_vector_type(4)))  int    v4i;

// Address-space qualified int4 pointers for the async-to-LDS builtin.
typedef __attribute__((address_space(1))) v4i* gptr_v4i;
typedef __attribute__((address_space(3))) v4i* lptr_v4i;

#define K_DIM 4096
#define N_DIM 4096
#define M_DIM 8192   // 4 * 2048

#if __has_builtin(__builtin_amdgcn_global_load_async_to_lds_b128)
#define USE_ASYNC 1
#else
#define USE_ASYNC 0
#endif

__constant__ float NF4_TAB[16] = {
    -1.0f, -0.6961928009986877f, -0.5250730514526367f, -0.39491748809814453f,
    -0.28444138169288635f, -0.18477343022823334f, -0.09105003625154495f, 0.0f,
    0.07958029955625534f, 0.16093020141124725f, 0.24611230194568634f,
    0.33791524171829224f, 0.44070982933044434f, 0.5626170039176941f,
    0.7229568362236023f, 1.0f};

// ---------------------------------------------------------------------------
// Kernel 1: NF4 dequant + transpose (through LDS, coalesced both sides).
//   q[b]: one byte as int32 (high nibble = value 2b, low = 2b+1).
//   flat value v = k*4096 + n ; block scale index = k*64 + n/64.
//   Output W_T[n][k] bf16 (N-major) so GEMM B tiles are K-contiguous.
// ---------------------------------------------------------------------------
__global__ __launch_bounds__(256)
void nf4_dequant_transpose(const int* __restrict__ q,
                           const float* __restrict__ scales,
                           bf16* __restrict__ wt) {
  __shared__ bf16 tile[64][36];            // [n_local][k_local], +4 pad
  const int t = threadIdx.x;
  const int kTile = blockIdx.x & 127;      // 128 tiles of 32 along K
  const int nTile = blockIdx.x >> 7;       // 64 tiles of 64 along N
  const int k0 = kTile * 32;
  const int n0 = nTile * 64;

  const int lv = t * 8;                    // linear value idx in tile
  const int kl = lv >> 6;                  // 0..31
  const int nl = lv & 63;                  // multiple of 8
  const int k  = k0 + kl;
  const float s = scales[k * 64 + (n0 >> 6)];
  const int byteIdx = k * 2048 + ((n0 + nl) >> 1);
#pragma unroll
  for (int b = 0; b < 4; ++b) {
    const int code = q[byteIdx + b];
    tile[nl + 2 * b + 0][kl] = (bf16)(NF4_TAB[(code >> 4) & 0xF] * s);
    tile[nl + 2 * b + 1][kl] = (bf16)(NF4_TAB[code & 0xF] * s);
  }
  __syncthreads();

  const int nr = t >> 2;                   // 0..63
  const int kc = (t & 3) * 8;              // 0,8,16,24
  v8bf o;
#pragma unroll
  for (int j = 0; j < 8; ++j) o[j] = tile[nr][kc + j];
  *(v8bf*)(wt + (size_t)(n0 + nr) * K_DIM + k0 + kc) = o;
}

// ---------------------------------------------------------------------------
// Kernel 2: x fp32 -> bf16 (elementwise, 8 per thread).
// ---------------------------------------------------------------------------
__global__ __launch_bounds__(256)
void f32_to_bf16(const float* __restrict__ x, bf16* __restrict__ xb) {
  const size_t i = ((size_t)blockIdx.x * 256 + threadIdx.x) * 8;
  v4f a = *(const v4f*)(x + i);
  v4f b = *(const v4f*)(x + i + 4);
  v8bf o;
#pragma unroll
  for (int j = 0; j < 4; ++j) { o[j] = (bf16)a[j]; o[j + 4] = (bf16)b[j]; }
  *(v8bf*)(xb + i) = o;
}

// ---------------------------------------------------------------------------
// Kernel 3: bf16 WMMA GEMM, C = A @ Bt^T + bias.
//   A  : [M][K] bf16 row-major (x)
//   Bt : [N][K] bf16 row-major (W transposed -> K-contiguous)
// Tile: BM=BN=128, BK=32; 256 threads = 8 wave32s (2 M x 4 N); each wave does
// a 64x32 sub-tile = 4x2 v_wmma_f32_16x16x32_bf16 per K-step.
// LDS is double-buffered; tiles stream in via async-to-LDS when available.
// ---------------------------------------------------------------------------
#define BM 128
#define BN 128
#define BK 32
#define LDS_K 40            // 32 + 8 pad, bf16 elements
#define TILE_E (BM * LDS_K) // elements per A/B tile buffer

#if USE_ASYNC
template <int N>
__device__ __forceinline__ void wait_async() {
#if __has_builtin(__builtin_amdgcn_s_wait_asynccnt)
  __builtin_amdgcn_s_wait_asynccnt(N);
#else
  asm volatile("s_wait_asynccnt %0" :: "i"(N) : "memory");
#endif
}
#endif

// Stage this thread's 2x16B of A and 2x16B of B into LDS.
__device__ __forceinline__ void stage_tile(const bf16* __restrict__ ag,
                                           const bf16* __restrict__ bg,
                                           bf16* asl, bf16* bsl) {
#if USE_ASYNC
  __builtin_amdgcn_global_load_async_to_lds_b128(
      (gptr_v4i)(ag),     (lptr_v4i)(asl),     0, 0);
  __builtin_amdgcn_global_load_async_to_lds_b128(
      (gptr_v4i)(ag + 8), (lptr_v4i)(asl + 8), 0, 0);
  __builtin_amdgcn_global_load_async_to_lds_b128(
      (gptr_v4i)(bg),     (lptr_v4i)(bsl),     0, 0);
  __builtin_amdgcn_global_load_async_to_lds_b128(
      (gptr_v4i)(bg + 8), (lptr_v4i)(bsl + 8), 0, 0);
#else
  v8bf a0 = *(const v8bf*)(ag);
  v8bf a1 = *(const v8bf*)(ag + 8);
  v8bf b0 = *(const v8bf*)(bg);
  v8bf b1 = *(const v8bf*)(bg + 8);
  *(v8bf*)(asl)     = a0;
  *(v8bf*)(asl + 8) = a1;
  *(v8bf*)(bsl)     = b0;
  *(v8bf*)(bsl + 8) = b1;
#endif
}

__global__ __launch_bounds__(256)
void gemm_bf16_wmma(const bf16* __restrict__ A,
                    const bf16* __restrict__ Bt,
                    const float* __restrict__ bias,
                    float* __restrict__ C) {
  __shared__ bf16 As[2 * TILE_E];
  __shared__ bf16 Bs[2 * TILE_E];

  const int t    = threadIdx.x;
  const int lane = t & 31;
  const int wid  = t >> 5;            // 0..7
  const int wm   = (wid >> 2) * 64;   // wave M offset: 0 / 64
  const int wn   = (wid & 3) * 32;    // wave N offset: 0..96
  const int m0   = blockIdx.y * BM;
  const int n0   = blockIdx.x * BN;

  // Staging assignment: thread -> (row, 16-col half); 16 bf16 per thread.
  const int lr = t >> 1;              // 0..127
  const int lh = (t & 1) * 16;        // 0 / 16
  const bf16* aptr = A  + (size_t)(m0 + lr) * K_DIM + lh;
  const bf16* bptr = Bt + (size_t)(n0 + lr) * K_DIM + lh;
  const int soff = lr * LDS_K + lh;   // LDS element offset for this thread

  const int l16 = lane & 15;
  const int hi  = lane >> 4;          // lane half select

  v8f acc[4][2] = {};

  constexpr int NITER = K_DIM / BK;   // 128

  // Prologue: stage tile 0 into buffer 0.
  stage_tile(aptr, bptr, As + soff, Bs + soff);

  for (int i = 0; i < NITER; ++i) {
    const int cur = i & 1;
    const bf16* as = As + cur * TILE_E;
    const bf16* bs = Bs + cur * TILE_E;

    // Issue next tile's loads into the other buffer (overlaps with compute).
    if (i + 1 < NITER) {
      const int nxt = cur ^ 1;
      stage_tile(aptr + (i + 1) * BK, bptr + (i + 1) * BK,
                 (bf16*)As + nxt * TILE_E + soff, (bf16*)Bs + nxt * TILE_E + soff);
#if USE_ASYNC
      wait_async<4>();   // current buffer's 4 loads done; next 4 in flight
#endif
    } else {
#if USE_ASYNC
      wait_async<0>();
#endif
    }
    __syncthreads();     // current buffer visible to all waves

    // ---- A fragments: lanes 0-15 hold K {0-7,16-23}; 16-31 hold {8-15,24-31}
    v16bf afrag[4];
#pragma unroll
    for (int mt = 0; mt < 4; ++mt) {
      const bf16* p = as + (wm + mt * 16 + l16) * LDS_K + hi * 8;
      v8bf lo = *(const v8bf*)p;
      v8bf hh = *(const v8bf*)(p + 16);
      afrag[mt] = __builtin_shufflevector(lo, hh, 0, 1, 2, 3, 4, 5, 6, 7,
                                          8, 9, 10, 11, 12, 13, 14, 15);
    }
    // ---- B fragments: lanes 0-15 hold K {0-7,16-23}; 16-31 hold {8-15,24-31}
    v16bf bfrag[2];
#pragma unroll
    for (int nt = 0; nt < 2; ++nt) {
      const bf16* p = bs + (wn + nt * 16 + l16) * LDS_K + hi * 8;
      v8bf lo = *(const v8bf*)p;
      v8bf hh = *(const v8bf*)(p + 16);
      bfrag[nt] = __builtin_shufflevector(lo, hh, 0, 1, 2, 3, 4, 5, 6, 7,
                                          8, 9, 10, 11, 12, 13, 14, 15);
    }

    // ---- 8 WMMAs per wave per K-step ----
#pragma unroll
    for (int mt = 0; mt < 4; ++mt)
#pragma unroll
      for (int nt = 0; nt < 2; ++nt)
        acc[mt][nt] = __builtin_amdgcn_wmma_f32_16x16x32_bf16(
            false, afrag[mt], false, bfrag[nt],
            (short)0, acc[mt][nt], false, false);

    __syncthreads();     // all waves done reading `cur` before it is restaged
  }

  // ---- epilogue: C/D layout -> rows r (lanes 0-15) and r+8 (lanes 16-31)
#pragma unroll
  for (int nt = 0; nt < 2; ++nt) {
    const int n = n0 + wn + nt * 16 + l16;
    const float bv = bias[n];
#pragma unroll
    for (int mt = 0; mt < 4; ++mt) {
#pragma unroll
      for (int r = 0; r < 8; ++r) {
        const int m = m0 + wm + mt * 16 + hi * 8 + r;
        C[(size_t)m * N_DIM + n] = acc[mt][nt][r] + bv;
      }
    }
  }
}

// ---------------------------------------------------------------------------
// Launch: dequant -> x convert -> GEMM (all on `stream`). ws layout:
//   [0, 32MiB)   W_T bf16 [4096][4096]
//   [32, 96MiB)  x  bf16 [8192][4096]
// ---------------------------------------------------------------------------
extern "C" void kernel_launch(void* const* d_in, const int* in_sizes, int n_in,
                              void* d_out, int out_size, void* d_ws, size_t ws_size,
                              hipStream_t stream) {
  const float* x      = (const float*)d_in[0];
  const int*   q      = (const int*)d_in[1];
  const float* scales = (const float*)d_in[2];
  const float* bias   = (const float*)d_in[3];
  float* out = (float*)d_out;

  bf16* wt = (bf16*)d_ws;
  bf16* xb = (bf16*)((char*)d_ws + (size_t)N_DIM * K_DIM * sizeof(bf16));

  nf4_dequant_transpose<<<128 * 64, 256, 0, stream>>>(q, scales, wt);
  f32_to_bf16<<<(M_DIM * (size_t)K_DIM) / (8 * 256), 256, 0, stream>>>(x, xb);
  gemm_bf16_wmma<<<dim3(N_DIM / BN, M_DIM / BM), 256, 0, stream>>>(xb, wt, bias, out);
}
